// SnapshotRetrieval_41463614275970
// MI455X (gfx1250) — matrix operand order repaired and checked by reading, not compiled, not run on hardware
//
#include <hip/hip_runtime.h>
#include <hip/hip_bf16.h>
#include <math.h>

// ---------------------------------------------------------------------------
// Shapes (fixed by reference setup_inputs)
// ---------------------------------------------------------------------------
#define Bc   2
#define Tc   2048
#define Dc   2048
#define Hc   16
#define DHc  64
#define Nc   512
#define MLc  128
#define BTc  (Bc * Tc)        // 4096
#define HDc  (Hc * DHc)       // 1024
#define KT   32               // K step per WMMA stage

typedef _Float16 v16h __attribute__((ext_vector_type(16)));
typedef float    v8f  __attribute__((ext_vector_type(8)));

// ---------------------------------------------------------------------------
// CDNA5 async global->LDS copy (16B per lane), ASYNCcnt-tracked.
// VDST = per-lane LDS byte address, VADDR = per-lane 64-bit global address.
// ---------------------------------------------------------------------------
__device__ __forceinline__ void async_copy_b128(unsigned lds_off, const float* gptr)
{
    asm volatile("global_load_async_to_lds_b128 %0, %1, off"
                 :: "v"(lds_off), "v"(gptr)
                 : "memory");
}
__device__ __forceinline__ void wait_async0()
{
    asm volatile("s_wait_asynccnt 0" ::: "memory");
}

// ---------------------------------------------------------------------------
// LDS-staged f32 GEMM on the WMMA pipe (f16 in, f32 accumulate).
// Block = 128 threads (4 waves), computes a 64x64 tile of C = A[M,K]@B[K,N].
// Wave w owns rows [w*16, w*16+16) and 4 accumulators covering 64 columns,
// so each A fragment feeds 4 v_wmma_f32_16x16x32_f16 per K-step.
// Fragment layouts per CDNA5 ISA 7.12.2 (16-bit A 16x32, B 32x16, C 16x16).
// Requires M%64==0, N%64==0, K%32==0 (true for all uses here).
// ---------------------------------------------------------------------------
__global__ void __launch_bounds__(128)
gemm_wmma_f16(const float* __restrict__ A,
              const float* __restrict__ B,
              float* __restrict__ C,
              int M, int K, int N)
{
    __shared__ float sA[64 * KT];   // [row 0..63][k 0..31]
    __shared__ float sB[KT * 64];   // [k 0..31][col 0..63]

    const int tid  = threadIdx.x;
    const int lane = tid & 31;
    const int wave = tid >> 5;

    const int ntb = N >> 6;                       // 64-wide column tiles
    const int tm0 = (blockIdx.x / ntb) << 6;
    const int tn0 = (blockIdx.x % ntb) << 6;

    const int half = lane >> 4;                   // 0: lanes 0-15, 1: 16-31
    const int lid  = lane & 15;

    const unsigned sA_base = (unsigned)(size_t)(void*)sA;   // LDS byte offset
    const unsigned sB_base = (unsigned)(size_t)(void*)sB;

    v8f acc0 = {}, acc1 = {}, acc2 = {}, acc3 = {};

    for (int k0 = 0; k0 < K; k0 += KT) {
        // ---- async-stage A (64x32 f32) and B (32x64 f32) tiles into LDS ----
#pragma unroll
        for (int it = 0; it < 4; ++it) {
            const int l = it * 128 + tid;         // 0..511, one b128 each
            {   // A: 8 chunks of 4 floats per row
                const int r = l >> 3, cc = (l & 7) << 2;
                const float* gp = A + (size_t)(tm0 + r) * K + k0 + cc;
                async_copy_b128(sA_base + (unsigned)((r * KT + cc) << 2), gp);
            }
            {   // B: 16 chunks of 4 floats per row
                const int r = l >> 4, cc = (l & 15) << 2;
                const float* gp = B + (size_t)(k0 + r) * N + tn0 + cc;
                async_copy_b128(sB_base + (unsigned)((r * 64 + cc) << 2), gp);
            }
        }
        wait_async0();
        __syncthreads();

        // ---- build fragments from LDS, 4 WMMAs per wave ----
        v16h a;
        const int ar = (wave << 4) + lid;
#pragma unroll
        for (int j = 0; j < 8; ++j) {
            a[j]     = (_Float16)sA[ar * KT + 8 * half + j];
            a[j + 8] = (_Float16)sA[ar * KT + 8 * half + j + 16];
        }
#pragma unroll
        for (int nt = 0; nt < 4; ++nt) {
            v16h b;
#pragma unroll
            for (int j = 0; j < 16; ++j)
                b[j] = (_Float16)sB[(8 * half + (j & 7) + ((j >> 3) << 4)) * 64 + nt * 16 + lid];
            v8f* accp = (nt == 0) ? &acc0 : (nt == 1) ? &acc1
                       : (nt == 2) ? &acc2 : &acc3;
            *accp = __builtin_amdgcn_wmma_f32_16x16x32_f16(
                        false, a, false, b, (short)0, *accp, false, false);
        }
        __syncthreads();                          // LDS reuse next K-step
    }

    // ---- store C: vgpr j -> row (8*half + j), col lid ----
    const int row0 = tm0 + (wave << 4) + (half << 3);
#pragma unroll
    for (int nt = 0; nt < 4; ++nt) {
        const v8f acc = (nt == 0) ? acc0 : (nt == 1) ? acc1
                      : (nt == 2) ? acc2 : acc3;
        float* cp = C + (size_t)row0 * N + tn0 + nt * 16 + lid;
#pragma unroll
        for (int j = 0; j < 8; ++j)
            cp[(size_t)j * N] = acc[j];
    }
}

// ---------------------------------------------------------------------------
// RMSNorm + full-dim RoPE + transpose for Q.
// grid (T, H, B), block 32. lane i owns elements (i, i+32) -> rope pair.
//   q_lin : [B, T, H*Dh]  ->  q_att : [B, H, T, Dh]
// ---------------------------------------------------------------------------
__global__ void qprep(const float* __restrict__ q_lin,
                      const float* __restrict__ scale,
                      const int*   __restrict__ token_offset,
                      float* __restrict__ q_att)
{
    const int t = blockIdx.x, h = blockIdx.y, b = blockIdx.z;
    const int i = threadIdx.x;

    const float* src = q_lin + (size_t)(b * Tc + t) * HDc + h * DHc;
    float x0 = src[i], x1 = src[i + 32];

    float ss = x0 * x0 + x1 * x1;
#pragma unroll
    for (int m = 16; m >= 1; m >>= 1) ss += __shfl_xor(ss, m, 32);
    const float r = rsqrtf(ss * (1.0f / 64.0f) + 1e-6f);
    x0 *= r * scale[i];
    x1 *= r * scale[i + 32];

    const float pos  = (float)(token_offset[0] + t);
    const float freq = powf(10000.0f, -(float)(2 * i) / 64.0f);
    const float ang  = pos * freq;
    const float c = cosf(ang), s = sinf(ang);

    float* dst = q_att + ((size_t)(b * Hc + h) * Tc + t) * DHc;
    dst[i]      = x0 * c - x1 * s;
    dst[i + 32] = x1 * c + x0 * s;
}

// ---------------------------------------------------------------------------
// K: RMSNorm + RoPE(snap_positions) + transpose; V: transpose only.
// grid (N, H, B), block 32.
//   k_lin/v_lin : [B, N, H, Dh]  ->  k_att/v_att : [B, H, N, Dh]
// ---------------------------------------------------------------------------
__global__ void kvprep(const float* __restrict__ k_lin,
                       const float* __restrict__ v_lin,
                       const float* __restrict__ kscale,
                       const int*   __restrict__ snap_pos,
                       float* __restrict__ k_att,
                       float* __restrict__ v_att)
{
    const int n = blockIdx.x, h = blockIdx.y, b = blockIdx.z;
    const int i = threadIdx.x;

    const size_t so = ((size_t)(b * Nc + n) * Hc + h) * DHc;
    const float* ks = k_lin + so;
    const float* vs = v_lin + so;

    float x0 = ks[i], x1 = ks[i + 32];
    float ss = x0 * x0 + x1 * x1;
#pragma unroll
    for (int m = 16; m >= 1; m >>= 1) ss += __shfl_xor(ss, m, 32);
    const float r = rsqrtf(ss * (1.0f / 64.0f) + 1e-6f);
    x0 *= r * kscale[i];
    x1 *= r * kscale[i + 32];

    const float pos  = (float)snap_pos[n];
    const float freq = powf(10000.0f, -(float)(2 * i) / 64.0f);
    const float ang  = pos * freq;
    const float c = cosf(ang), s = sinf(ang);

    const size_t dof = ((size_t)(b * Hc + h) * Nc + n) * DHc;
    float* kd = k_att + dof;
    float* vd = v_att + dof;
    kd[i]      = x0 * c - x1 * s;
    kd[i + 32] = x1 * c + x0 * s;
    vd[i]      = vs[i];
    vd[i + 32] = vs[i + 32];
}

// ---------------------------------------------------------------------------
// Attention row kernel: grid (T, H, B), 512 threads (one per snapshot n).
// scores -> causal mask -> radix-select 64th largest -> sink softmax ->
// weights @ V -> sigmoid gate -> att_out [B, T, H*Dh]
// ---------------------------------------------------------------------------
__global__ void attn(const float* __restrict__ q_att,
                     const float* __restrict__ k_att,
                     const float* __restrict__ v_att,
                     const float* __restrict__ gate_lin,
                     const float* __restrict__ sink_logit,
                     const int*   __restrict__ snap_pos,
                     const int*   __restrict__ token_offset,
                     float* __restrict__ att_out)
{
    const int n = threadIdx.x;
    const int t = blockIdx.x, h = blockIdx.y, b = blockIdx.z;

    __shared__ float    qs[DHc];
    __shared__ float    sw[Nc];
    __shared__ float    red[Nc];
    __shared__ unsigned scnt;

    const float* qrow = q_att + ((size_t)(b * Hc + h) * Tc + t) * DHc;
    if (n < DHc) qs[n] = qrow[n];
    __syncthreads();

    // score
    const float* krow = k_att + ((size_t)(b * Hc + h) * Nc + n) * DHc;
    float s = 0.f;
#pragma unroll
    for (int d = 0; d < DHc; ++d) s += qs[d] * krow[d];
    s *= 0.125f;                                  // 1/sqrt(64)
    const int t_abs = token_offset[0] + t;
    if (!(snap_pos[n] < t_abs)) s = -3.0e38f;     // causal mask

    // radix select of the 64th-largest value (monotone float->uint map)
    unsigned key = __float_as_uint(s);
    key = (key & 0x80000000u) ? ~key : (key | 0x80000000u);
    unsigned prefix = 0;
    for (int bit = 31; bit >= 0; --bit) {
        if (n == 0) scnt = 0;
        __syncthreads();
        const unsigned cand = prefix | (1u << bit);
        if ((key >> bit) >= (cand >> bit)) atomicAdd(&scnt, 1u);
        __syncthreads();
        if (scnt >= 64u) prefix = cand;           // uniform decision
        __syncthreads();
    }
    const bool  sel  = (key >= prefix);
    const float ssel = sel ? s : -3.0e38f;

    // max (including sink)
    red[n] = ssel;
    __syncthreads();
    for (int st = 256; st > 0; st >>= 1) {
        if (n < st) red[n] = fmaxf(red[n], red[n + st]);
        __syncthreads();
    }
    const float sink = sink_logit[h];
    const float m    = fmaxf(red[0], sink);
    __syncthreads();

    // exp & sum
    const float e = expf(ssel - m);
    sw[n]  = e;
    red[n] = e;
    __syncthreads();
    for (int st = 256; st > 0; st >>= 1) {
        if (n < st) red[n] += red[n + st];
        __syncthreads();
    }
    const float inv = 1.0f / (red[0] + expf(sink - m));
    __syncthreads();

    // out[d] = inv * sum_n e[n] * V[n, d]   (8 groups x 64 d)
    const int d = n & 63;
    const int g = n >> 6;
    const float* vbase = v_att + ((size_t)(b * Hc + h) * Nc) * DHc;
    float acc = 0.f;
#pragma unroll 8
    for (int j = 0; j < 64; ++j) {
        const int nn = g * 64 + j;
        acc += sw[nn] * vbase[(size_t)nn * DHc + d];
    }
    red[n] = acc;
    __syncthreads();
    if (n < DHc) {
        float o = 0.f;
#pragma unroll
        for (int gg = 0; gg < 8; ++gg) o += red[gg * 64 + n];
        o *= inv;
        const size_t oi = (size_t)(b * Tc + t) * HDc + h * DHc + n;
        const float gl = gate_lin[oi];
        o *= 1.0f / (1.0f + expf(-gl));           // sigmoid gate
        att_out[oi] = o;
    }
}

// ---------------------------------------------------------------------------
// Launch
// ---------------------------------------------------------------------------
static inline void launch_gemm(const float* A, const float* B, float* C,
                               int M, int K, int N, hipStream_t stream)
{
    const int blocks = (M / 64) * (N / 64);       // one 64x64 tile per block
    gemm_wmma_f16<<<blocks, 128, 0, stream>>>(A, B, C, M, K, N);
}

extern "C" void kernel_launch(void* const* d_in, const int* in_sizes, int n_in,
                              void* d_out, int out_size, void* d_ws, size_t ws_size,
                              hipStream_t stream)
{
    (void)in_sizes; (void)n_in; (void)out_size; (void)ws_size;

    const float* x       = (const float*)d_in[0];
    const float* snaps   = (const float*)d_in[1];   // [B,N,H,R] == [B*N*H, 64]
    const float* q_down  = (const float*)d_in[2];
    const float* q_up    = (const float*)d_in[3];
    const float* g_down  = (const float*)d_in[4];
    const float* g_up    = (const float*)d_in[5];
    const float* o_down  = (const float*)d_in[6];
    const float* o_up    = (const float*)d_in[7];
    const float* k_up    = (const float*)d_in[8];
    const float* v_up    = (const float*)d_in[9];
    const float* q_scale = (const float*)d_in[10];
    const float* k_scale = (const float*)d_in[11];
    const float* sink    = (const float*)d_in[12];
    const int*   spos    = (const int*)d_in[13];
    const int*   toff    = (const int*)d_in[14];
    float* out = (float*)d_out;

    // workspace carve-up (floats)
    float* ws    = (float*)d_ws;
    float* q_mid = ws;                                  // [4096,128]
    float* g_mid = q_mid + (size_t)BTc * MLc;           // [4096,128]
    float* q_lin = g_mid + (size_t)BTc * MLc;           // [4096,1024]
    float* g_lin = q_lin + (size_t)BTc * HDc;           // [4096,1024]
    float* q_att = g_lin + (size_t)BTc * HDc;           // [B,H,T,Dh]
    float* k_lin = q_att + (size_t)BTc * HDc;           // [B,N,H,Dh]
    float* v_lin = k_lin + (size_t)Bc * Nc * Hc * DHc;
    float* k_att = v_lin + (size_t)Bc * Nc * Hc * DHc;  // [B,H,N,Dh]
    float* v_att = k_att + (size_t)Bc * Nc * Hc * DHc;
    float* a_out = v_att + (size_t)Bc * Nc * Hc * DHc;  // [4096,1024] gated
    float* o_mid = a_out + (size_t)BTc * HDc;           // [4096,128]

    // dense projections (WMMA path, async-LDS staged)
    launch_gemm(x,     q_down, q_mid, BTc,          Dc,  MLc, stream);
    launch_gemm(x,     g_down, g_mid, BTc,          Dc,  MLc, stream);
    launch_gemm(q_mid, q_up,   q_lin, BTc,          MLc, HDc, stream);
    launch_gemm(g_mid, g_up,   g_lin, BTc,          MLc, HDc, stream);
    launch_gemm(snaps, k_up,   k_lin, Bc * Nc * Hc, DHc, DHc, stream);
    launch_gemm(snaps, v_up,   v_lin, Bc * Nc * Hc, DHc, DHc, stream);

    // norm + rope + transpose
    qprep <<<dim3(Tc, Hc, Bc), 32, 0, stream>>>(q_lin, q_scale, toff, q_att);
    kvprep<<<dim3(Nc, Hc, Bc), 32, 0, stream>>>(k_lin, v_lin, k_scale, spos,
                                                k_att, v_att);

    // top-64 sink-softmax attention + gate
    attn<<<dim3(Tc, Hc, Bc), Nc, 0, stream>>>(q_att, k_att, v_att, g_lin,
                                              sink, spos, toff, a_out);

    // output projection (WMMA path)
    launch_gemm(a_out, o_down, o_mid, BTc, HDc, MLc, stream);
    launch_gemm(o_mid, o_up,   out,   BTc, MLc, Dc,  stream);
}
